// DictionaryModule_39015482917650
// MI455X (gfx1250) — matrix operand order repaired and checked by reading, not compiled, not run on hardware
//
#include <hip/hip_runtime.h>
#include <hip/hip_bf16.h>

typedef __attribute__((ext_vector_type(16))) __bf16 v16bf;
typedef __attribute__((ext_vector_type(8)))  __bf16 v8bf;
typedef __attribute__((ext_vector_type(8)))  float  v8f;
typedef int v4i __attribute__((vector_size(16)));

#define FEAT 1024
#define NQ   16384
#define DICT 8192
#define TEMP 0.15f

constexpr int LDSTR = 56;   // padded LDS row stride (bf16 elems); 112B = 16B-aligned, conflict-free frag reads

// ---------------------------------------------------------------------------
// CDNA5 async global->LDS copy (ASYNCcnt path), with sync fallback
// ---------------------------------------------------------------------------
#if defined(__has_builtin)
#if __has_builtin(__builtin_amdgcn_global_load_async_to_lds_b128)
#define ASYNC_LDS 1
#else
#define ASYNC_LDS 0
#endif
#else
#define ASYNC_LDS 0
#endif

__device__ inline void cp_async16(__bf16* ldst, const __bf16* gsrc) {
#if ASYNC_LDS
    __builtin_amdgcn_global_load_async_to_lds_b128(
        (__attribute__((address_space(1))) v4i*)(void*)gsrc,
        (__attribute__((address_space(3))) v4i*)(void*)ldst, 0, 0);
#else
    *(int4*)ldst = *(const int4*)gsrc;
#endif
}

__device__ inline void async_fence() {
#if ASYNC_LDS
#if __has_builtin(__builtin_amdgcn_s_wait_asynccnt)
    __builtin_amdgcn_s_wait_asynccnt(0);
#else
    asm volatile("s_wait_asynccnt 0" ::: "memory");
#endif
#endif
}

// ---------------------------------------------------------------------------
// W [dim][dim] f32 -> Wt [dim][dim] bf16 (transposed), classic LDS tile
// ---------------------------------------------------------------------------
__global__ __launch_bounds__(1024)
void transpose_to_bf16(const float* __restrict__ in, __bf16* __restrict__ out, int dim) {
    __shared__ float tile[32][33];
    int x = blockIdx.x * 32 + threadIdx.x;
    int y = blockIdx.y * 32 + threadIdx.y;
    tile[threadIdx.y][threadIdx.x] = in[(size_t)y * dim + x];
    __syncthreads();
    int ox = blockIdx.y * 32 + threadIdx.x;
    int oy = blockIdx.x * 32 + threadIdx.y;
    out[(size_t)oy * dim + ox] = (__bf16)tile[threadIdx.x][threadIdx.y];
}

// ---------------------------------------------------------------------------
// f32 rows -> bf16 rows + reciprocal L2 norm per row (matches F.normalize eps)
// ---------------------------------------------------------------------------
__global__ __launch_bounds__(256)
void convert_rows_rnorm(const float* __restrict__ in, __bf16* __restrict__ outb,
                        float* __restrict__ rnorm, int cols) {
    int row = blockIdx.x, tid = threadIdx.x;
    const float* src = in + (size_t)row * cols;
    float s = 0.f;
    for (int c = tid; c < cols; c += 256) {
        float v = src[c];
        s += v * v;
        outb[(size_t)row * cols + c] = (__bf16)v;
    }
    __shared__ float red[256];
    red[tid] = s; __syncthreads();
    for (int st = 128; st > 0; st >>= 1) {
        if (tid < st) red[tid] += red[tid + st];
        __syncthreads();
    }
    if (tid == 0) rnorm[row] = 1.0f / fmaxf(sqrtf(red[0]), 1e-12f);
}

__global__ __launch_bounds__(256)
void rnorm_rows_bf16(const __bf16* __restrict__ in, float* __restrict__ rnorm, int cols) {
    int row = blockIdx.x, tid = threadIdx.x;
    const __bf16* src = in + (size_t)row * cols;
    float s = 0.f;
    for (int c = tid; c < cols; c += 256) {
        float v = (float)src[c];
        s += v * v;
    }
    __shared__ float red[256];
    red[tid] = s; __syncthreads();
    for (int st = 128; st > 0; st >>= 1) {
        if (tid < st) red[tid] += red[tid + st];
        __syncthreads();
    }
    if (tid == 0) rnorm[row] = 1.0f / fmaxf(sqrtf(red[0]), 1e-12f);
}

// ---------------------------------------------------------------------------
// WMMA fragment loads from LDS (layouts per CDNA5 ISA 7.12.2, wave32)
// A 16x32 bf16: lane m=l&15, contiguous K chunks at kh and 16+kh (kh=(l>>4)*8)
// B 32x16 bf16: lane n=l&15, contiguous K chunks at kh and 16+kh (kh=(l>>4)*8)
// ---------------------------------------------------------------------------
__device__ inline v16bf ld_a_frag(const __bf16* tile, int m0, int lane) {
    int m  = m0 + (lane & 15);
    int kh = (lane >> 4) * 8;
    const __bf16* r = tile + m * LDSTR;
    union { v16bf v; v8bf h[2]; } u;
    u.h[0] = *(const v8bf*)(r + kh);
    u.h[1] = *(const v8bf*)(r + 16 + kh);
    return u.v;
}

__device__ inline v16bf ld_b_frag(const __bf16* tile, int n0, int lane) {
    int n  = n0 + (lane & 15);
    int kh = (lane >> 4) * 8;
    const __bf16* r = tile + n * LDSTR;
    union { v16bf v; v8bf h[2]; } u;
    u.h[0] = *(const v8bf*)(r + kh);
    u.h[1] = *(const v8bf*)(r + 16 + kh);
    return u.v;
}

// ---------------------------------------------------------------------------
// NT GEMM: C[M,N] = A[M,K] * B[N,K]^T, 128x128 tile, 8 waves of wave32,
// double-buffered LDS (async global->LDS staging), v_wmma_f32_16x16x32_bf16.
// EPI 0: relu(acc + bias[col]) -> bf16     (aux1 = bias)
// EPI 1: acc + bias[col]       -> bf16     (aux1 = bias)
// EPI 2: acc * aux1[row] * aux2[col] -> f32 (cosine sim epilogue)
// ---------------------------------------------------------------------------
template <int EPI, bool AF32>
__global__ __launch_bounds__(256)
void gemm_nt(const void* __restrict__ Ap, const __bf16* __restrict__ B,
             void* __restrict__ Cp, const float* __restrict__ aux1,
             const float* __restrict__ aux2, int N, int K) {
    __shared__ __bf16 lA[2][128 * LDSTR];
    __shared__ __bf16 lB[2][128 * LDSTR];

    const int tid   = threadIdx.x;
    const int lane  = tid & 31;
    const int wave  = tid >> 5;
    const int m_blk = blockIdx.y * 128;
    const int n_blk = blockIdx.x * 128;

    const int srow = tid >> 1;         // staging: 0..127
    const int scol = (tid & 1) * 16;   // 0 or 16

    const float*  Af = (const float*)Ap;
    const __bf16* Ab = (const __bf16*)Ap;

    auto stage = [&](int buf, int k0) {
        // ---- A tile (128 x 32) ----
        if constexpr (AF32) {
            // f32 -> bf16 convert in registers, then ds_store
            __bf16 tmp[16];
            const float* src = Af + (size_t)(m_blk + srow) * K + k0 + scol;
#pragma unroll
            for (int i = 0; i < 16; i += 4) {
                float4 f = *(const float4*)(src + i);
                tmp[i + 0] = (__bf16)f.x; tmp[i + 1] = (__bf16)f.y;
                tmp[i + 2] = (__bf16)f.z; tmp[i + 3] = (__bf16)f.w;
            }
            __bf16* dstA = &lA[buf][srow * LDSTR + scol];
            *(int4*)(dstA)     = *(int4*)(tmp);
            *((int4*)dstA + 1) = *((int4*)tmp + 1);
        } else {
            // async global->LDS (ASYNCcnt), bypasses VGPRs
            const __bf16* src = Ab + (size_t)(m_blk + srow) * K + k0 + scol;
            __bf16* dstA = &lA[buf][srow * LDSTR + scol];
            cp_async16(dstA,     src);
            cp_async16(dstA + 8, src + 8);
        }
        // ---- B tile (128 x 32), always bf16 [N][K] row-major ----
        const __bf16* srcB = B + (size_t)(n_blk + srow) * K + k0 + scol;
        __bf16* dstB = &lB[buf][srow * LDSTR + scol];
        cp_async16(dstB,     srcB);
        cp_async16(dstB + 8, srcB + 8);
    };

    const int m_off = (wave >> 1) * 32;  // 0,32,64,96
    const int n_off = (wave & 1) * 64;   // 0,64

    v8f acc[2][4];
    v8f zero = {0.f, 0.f, 0.f, 0.f, 0.f, 0.f, 0.f, 0.f};
#pragma unroll
    for (int i = 0; i < 2; ++i)
#pragma unroll
        for (int j = 0; j < 4; ++j) acc[i][j] = zero;

    const int nk = K / 32;
    stage(0, 0);
    async_fence();
    __syncthreads();

    for (int ks = 0; ks < nk; ++ks) {
        const int buf = ks & 1;
        if (ks + 1 < nk) stage(buf ^ 1, (ks + 1) * 32);

        v16bf af[2], bf[4];
#pragma unroll
        for (int i = 0; i < 2; ++i) af[i] = ld_a_frag(lA[buf], m_off + i * 16, lane);
#pragma unroll
        for (int j = 0; j < 4; ++j) bf[j] = ld_b_frag(lB[buf], n_off + j * 16, lane);

#pragma unroll
        for (int i = 0; i < 2; ++i)
#pragma unroll
            for (int j = 0; j < 4; ++j)
                acc[i][j] = __builtin_amdgcn_wmma_f32_16x16x32_bf16(
                    false, af[i], false, bf[j], (short)0, acc[i][j], false, false);

        async_fence();       // this wave's async stores to LDS complete
        __syncthreads();     // publish staged buffer across all waves
    }

    // ---- epilogue: C layout per ISA (lane n=l&15; VGPR r -> row r + (l>>4)*8)
#pragma unroll
    for (int i = 0; i < 2; ++i) {
#pragma unroll
        for (int j = 0; j < 4; ++j) {
            int col   = n_blk + n_off + j * 16 + (lane & 15);
            int rbase = m_blk + m_off + i * 16 + (lane >> 4) * 8;
#pragma unroll
            for (int r = 0; r < 8; ++r) {
                int row = rbase + r;
                float v = acc[i][j][r];
                if constexpr (EPI == 0) {
                    v = fmaxf(v + aux1[col], 0.f);
                    ((__bf16*)Cp)[(size_t)row * N + col] = (__bf16)v;
                } else if constexpr (EPI == 1) {
                    v += aux1[col];
                    ((__bf16*)Cp)[(size_t)row * N + col] = (__bf16)v;
                } else {
                    v *= aux1[row] * aux2[col];
                    ((float*)Cp)[(size_t)row * N + col] = v;
                }
            }
        }
    }
}

// ---------------------------------------------------------------------------
// Per-row top-5 over 8192 sims -> softmax(T=0.15) -> weighted value gather
// ---------------------------------------------------------------------------
__device__ inline void merge5(float* av, int* ai, const float* bv, const int* bi) {
    float rv[5]; int ri[5];
    int i = 0, j = 0;
#pragma unroll
    for (int t = 0; t < 5; ++t) {
        if (j >= 5 || (i < 5 && av[i] >= bv[j])) { rv[t] = av[i]; ri[t] = ai[i]; ++i; }
        else                                     { rv[t] = bv[j]; ri[t] = bi[j]; ++j; }
    }
#pragma unroll
    for (int t = 0; t < 5; ++t) { av[t] = rv[t]; ai[t] = ri[t]; }
}

__global__ __launch_bounds__(256)
void topk_softmax_gather(const float* __restrict__ sim, const float* __restrict__ values,
                         float* __restrict__ out) {
    const int row = blockIdx.x, tid = threadIdx.x;
    const float* s = sim + (size_t)row * DICT;

    float bv[5]; int bi[5];
#pragma unroll
    for (int t = 0; t < 5; ++t) { bv[t] = -__builtin_inff(); bi[t] = 0; }

    for (int c = tid; c < DICT; c += 256) {
        float x = s[c];
        if (x > bv[4]) {
            int p = 4;
            while (p > 0 && x > bv[p - 1]) { bv[p] = bv[p - 1]; bi[p] = bi[p - 1]; --p; }
            bv[p] = x; bi[p] = c;
        }
    }

    __shared__ float sv[256 * 5];
    __shared__ int   si[256 * 5];
#pragma unroll
    for (int t = 0; t < 5; ++t) { sv[tid * 5 + t] = bv[t]; si[tid * 5 + t] = bi[t]; }

    for (int st = 128; st > 0; st >>= 1) {
        __syncthreads();
        if (tid < st) {
            float mv[5], ov[5]; int mi[5], oi[5];
#pragma unroll
            for (int t = 0; t < 5; ++t) {
                mv[t] = sv[tid * 5 + t];        mi[t] = si[tid * 5 + t];
                ov[t] = sv[(tid + st) * 5 + t]; oi[t] = si[(tid + st) * 5 + t];
            }
            merge5(mv, mi, ov, oi);
#pragma unroll
            for (int t = 0; t < 5; ++t) { sv[tid * 5 + t] = mv[t]; si[tid * 5 + t] = mi[t]; }
        }
    }
    __syncthreads();

    float tv[5]; int ti[5];
#pragma unroll
    for (int t = 0; t < 5; ++t) { tv[t] = sv[t]; ti[t] = si[t]; }
    float mx = tv[0];              // sorted descending
    float w[5], sum = 0.f;
#pragma unroll
    for (int t = 0; t < 5; ++t) { w[t] = __expf((tv[t] - mx) / TEMP); sum += w[t]; }
    float inv = 1.f / sum;

    for (int c = tid; c < FEAT; c += 256) {
        float o = 0.f;
#pragma unroll
        for (int t = 0; t < 5; ++t) o += w[t] * values[(size_t)ti[t] * FEAT + c];
        out[(size_t)row * FEAT + c] = o * inv;
    }
}

// ---------------------------------------------------------------------------
extern "C" void kernel_launch(void* const* d_in, const int* in_sizes, int n_in,
                              void* d_out, int out_size, void* d_ws, size_t ws_size,
                              hipStream_t stream) {
    (void)in_sizes; (void)n_in; (void)out_size; (void)ws_size;

    const float* q_feats = (const float*)d_in[0];
    const float* W1      = (const float*)d_in[1];
    const float* b1      = (const float*)d_in[2];
    const float* W2      = (const float*)d_in[3];
    const float* b2      = (const float*)d_in[4];
    const float* keys    = (const float*)d_in[5];
    const float* values  = (const float*)d_in[6];

    char* ws = (char*)d_ws;
    __bf16* W1t = (__bf16*)ws; ws += (size_t)FEAT * FEAT * 2;
    __bf16* W2t = (__bf16*)ws; ws += (size_t)FEAT * FEAT * 2;
    __bf16* hb  = (__bf16*)ws; ws += (size_t)NQ * FEAT * 2;
    __bf16* qb  = (__bf16*)ws; ws += (size_t)NQ * FEAT * 2;
    __bf16* kb  = (__bf16*)ws; ws += (size_t)DICT * FEAT * 2;
    float*  qrn = (float*)ws;  ws += (size_t)NQ * 4;
    float*  krn = (float*)ws;

    float* retrieved = (float*)d_out;
    float* sim       = retrieved + (size_t)NQ * FEAT;

    dim3 tb(32, 32);
    transpose_to_bf16<<<dim3(FEAT / 32, FEAT / 32), tb, 0, stream>>>(W1, W1t, FEAT);
    transpose_to_bf16<<<dim3(FEAT / 32, FEAT / 32), tb, 0, stream>>>(W2, W2t, FEAT);
    convert_rows_rnorm<<<DICT, 256, 0, stream>>>(keys, kb, krn, FEAT);

    // h = relu(q_feats @ W1 + b1)
    gemm_nt<0, true ><<<dim3(FEAT / 128, NQ / 128), 256, 0, stream>>>(q_feats, W1t, hb, b1, nullptr, FEAT, FEAT);
    // q_proj = h @ W2 + b2
    gemm_nt<1, false><<<dim3(FEAT / 128, NQ / 128), 256, 0, stream>>>(hb, W2t, qb, b2, nullptr, FEAT, FEAT);
    rnorm_rows_bf16<<<NQ, 256, 0, stream>>>(qb, qrn, FEAT);
    // sim = (q_proj @ keys^T) * (1/|q|) * (1/|k|)
    gemm_nt<2, false><<<dim3(DICT / 128, NQ / 128), 256, 0, stream>>>(qb, kb, sim, qrn, krn, DICT, FEAT);

    topk_softmax_gather<<<NQ, 256, 0, stream>>>(sim, values, retrieved);
}
